// TaskExperts_90615220011775
// MI455X (gfx1250) — matrix operand order repaired and checked by reading, not compiled, not run on hardware
//
#include <hip/hip_runtime.h>

// Problem constants (match reference)
#define T_TOK 4096
#define DH    1024
#define FI    4096
#define NE    8

// Tiling
#define TT 32     // tokens per block
#define FC 64     // F-chunk
#define KC 32     // K(D)-chunk for GEMM1 (= one WMMA K step)
#define DC 64     // D-chunk for GEMM2 staging

typedef __attribute__((ext_vector_type(16))) __bf16 v16bf;
typedef __attribute__((ext_vector_type(8)))  __bf16 v8bf;
typedef __attribute__((ext_vector_type(4)))  __bf16 v4bf;
typedef __attribute__((ext_vector_type(8)))  float  v8f;
typedef __attribute__((ext_vector_type(4)))  unsigned int u32x4;
typedef __attribute__((ext_vector_type(8)))  int   i32x8;
typedef __attribute__((ext_vector_type(4)))  int   i32x4;

union BF16x16 { struct { v8bf lo, hi; } h; v16bf v; };

__device__ __forceinline__ v8bf cvt8(float4 a, float4 b) {
  v8bf r;
  r[0] = (__bf16)a.x; r[1] = (__bf16)a.y; r[2] = (__bf16)a.z; r[3] = (__bf16)a.w;
  r[4] = (__bf16)b.x; r[5] = (__bf16)b.y; r[6] = (__bf16)b.z; r[7] = (__bf16)b.w;
  return r;
}

// Convert 16 consecutive fp32 (16B-aligned, LDS) into a v16bf WMMA operand.
__device__ __forceinline__ v16bf cvt16(const float* p) {
  const float4* q = (const float4*)p;
  BF16x16 r;
  r.h.lo = cvt8(q[0], q[1]);
  r.h.hi = cvt8(q[2], q[3]);
  return r.v;
}

// ---------------------------------------------------------------------------
// Tensor Data Mover: async DMA of a 2D fp32 tile (tile_x cols x tile_y rows)
// from global [tensor_y x tensor_x] (row stride stride_x elems) into LDS,
// with pad_amt(+1) dwords of LDS padding after every 2^(pad_iv+1) dwords.
// D# per CDNA5 ISA. Issued by one wave; TENSORcnt.
// ---------------------------------------------------------------------------
__device__ __forceinline__ void tdm_load_2d(unsigned lds_off, const float* gptr,
                                            unsigned tile_x, unsigned tile_y,
                                            unsigned tensor_x, unsigned tensor_y,
                                            unsigned stride_x,
                                            unsigned pad_iv, unsigned pad_amt) {
  unsigned long long ga = (unsigned long long)(size_t)gptr;
  u32x4 g0;
  g0[0] = 1u;                                             // count=1 (valid D#)
  g0[1] = lds_off;                                        // lds_addr
  g0[2] = (unsigned)(ga & 0xFFFFFFFFu);                   // global_addr[31:0]
  g0[3] = (unsigned)((ga >> 32) & 0x01FFFFFFu) | 0x80000000u; // [56:32] | type=2
  i32x8 g1;
  g1[0] = (int)((2u << 16) | (1u << 20) |                 // data_size=4B, pad_en
                (pad_iv << 22) | (pad_amt << 25));
  g1[1] = (int)((tensor_x & 0xFFFFu) << 16);              // tensor_dim0[15:0]
  g1[2] = (int)((tensor_x >> 16) | ((tensor_y & 0xFFFFu) << 16));
  g1[3] = (int)((tensor_y >> 16) | (tile_x << 16));       // tile_dim0
  g1[4] = (int)tile_y;                                    // tile_dim1 (dim2=0)
  g1[5] = (int)stride_x;                                  // tensor_dim0_stride lo
  g1[6] = 0;
  g1[7] = 0;
  i32x4 gz = {0, 0, 0, 0};
#if __clang_major__ >= 23
  i32x8 gz8 = {0, 0, 0, 0, 0, 0, 0, 0};
  __builtin_amdgcn_tensor_load_to_lds(g0, g1, gz, gz, gz8, 0);
#else
  __builtin_amdgcn_tensor_load_to_lds(g0, g1, gz, gz, 0);
#endif
}

__device__ __forceinline__ unsigned lds_off_of(const void* p) {
  return (unsigned)(size_t)p;   // addrspace(3) offset lives in low 32 bits
}

// ---------------------------------------------------------------------------
// Kernel 1: routing. logits = x@gate_w^T + gate_b, softmax->top2->renorm.
// Softmax denominator cancels under top-2 renorm: cw1 = 1/(1+exp(l2-l1)).
// ---------------------------------------------------------------------------
__global__ void __launch_bounds__(256)
moe_route(const float* __restrict__ x, const float* __restrict__ gw,
          const float* __restrict__ gb, float* __restrict__ cw) {
  __shared__ float gws[NE][DH];   // 32 KB
  const int tid = threadIdx.x;
  for (int i = tid; i < NE * DH; i += 256)
    gws[i >> 10][i & (DH - 1)] = gw[i];
  __syncthreads();

  const int t = blockIdx.x * 256 + tid;
  const float* xr = x + (size_t)t * DH;
  float acc[NE];
#pragma unroll
  for (int e = 0; e < NE; ++e) acc[e] = 0.f;

  for (int d = 0; d < DH; d += 4) {
    float4 xv = *(const float4*)(xr + d);
#pragma unroll
    for (int e = 0; e < NE; ++e) {
      acc[e] += xv.x * gws[e][d] + xv.y * gws[e][d + 1] +
                xv.z * gws[e][d + 2] + xv.w * gws[e][d + 3];
    }
  }
#pragma unroll
  for (int e = 0; e < NE; ++e) acc[e] += gb[e];

  int i1 = 0; float l1 = acc[0];
#pragma unroll
  for (int e = 1; e < NE; ++e) if (acc[e] > l1) { l1 = acc[e]; i1 = e; }
  int i2 = -1; float l2 = -3.4e38f;
#pragma unroll
  for (int e = 0; e < NE; ++e) if (e != i1 && acc[e] > l2) { l2 = acc[e]; i2 = e; }

  const float p1 = 1.f / (1.f + __expf(l2 - l1));
  const float p2 = 1.f - p1;

  float o[NE];
#pragma unroll
  for (int e = 0; e < NE; ++e) o[e] = 0.f;
  o[i1] = p1; o[i2] = p2;

  float4* op = (float4*)(cw + (size_t)t * NE);
  op[0] = make_float4(o[0], o[1], o[2], o[3]);
  op[1] = make_float4(o[4], o[5], o[6], o[7]);
}

// ---------------------------------------------------------------------------
// Kernel 2: fused dense-MoE SwiGLU FFN.
// Grid = T/TT blocks; expert loop inside so concurrent blocks stream the same
// expert weights through L2. Weight tiles arrive via double-buffered TDM
// (tensor_load_to_lds) as fp32 (+LDS row padding); fp32->bf16 happens at
// operand build. cw folded into act; fp32 out tile accumulated in LDS.
// ---------------------------------------------------------------------------
__global__ void __launch_bounds__(256)
moe_experts(const float* __restrict__ x,
            const float* __restrict__ w1, const float* __restrict__ w3,
            const float* __restrict__ w2, const float* __restrict__ cw,
            float* __restrict__ out) {
  __shared__ __align__(16) float  outs[TT][1028];        // 131,584 B
  __shared__ __align__(16) __bf16 xs [TT][KC + 16];      //   3,072 B
  __shared__ __align__(16) float  w1s[2][FC][KC + 4];    //  18,432 B (TDM, stride 36)
  __shared__ __align__(16) float  w3s[2][FC][KC + 4];    //  18,432 B
  __shared__ __align__(16) __bf16 acts[TT][FC + 8];      //   4,608 B
  __shared__ __align__(16) float  w2s[2][DC][FC + 4];    //  34,816 B (TDM, stride 68)
  __shared__ __align__(16) float  cws[TT][NE];           //   1,024 B

  const int tid = threadIdx.x;
  const int wv  = tid >> 5;          // wave id 0..7
  const int ln  = tid & 15;          // lane % 16
  const int hl  = (tid >> 4) & 1;    // lane / 16
  const int t0  = blockIdx.x * TT;

  for (int i = tid; i < TT * 1028; i += 256) ((float*)outs)[i] = 0.f;
  {
    const int r = tid >> 3, c = tid & 7;
    cws[r][c] = cw[(size_t)(t0 + r) * NE + c];
  }
  __syncthreads();

  const int mt = wv & 1;    // token sub-tile (2 x 16)
  const int nt = wv >> 1;   // f/d sub-tile   (4 x 16)

  for (int e = 0; e < NE; ++e) {
    const float* W1e = w1 + (size_t)e * FI * DH;
    const float* W3e = w3 + (size_t)e * FI * DH;
    const float* W2e = w2 + (size_t)e * DH * FI;

    for (int fc = 0; fc < FI; fc += FC) {
      v8f h1 = {}; v8f h3 = {};

      // prologue: TDM-load first w1/w3 tiles (64f x 32d fp32) into buffer 0
      if (wv == 0) {
        tdm_load_2d(lds_off_of(&w1s[0][0][0]), W1e + (size_t)fc * DH,
                    KC, FC, DH, FI, DH, 4, 3);
        tdm_load_2d(lds_off_of(&w3s[0][0][0]), W3e + (size_t)fc * DH,
                    KC, FC, DH, FI, DH, 4, 3);
      }

      int pb = 0;
      // ---- GEMM1: h1/h3[32 x 64] over K = D = 1024 ----
      for (int kc = 0; kc < DH; kc += KC) {
        // stage x tile 32x32 (f32 -> bf16), coalesced float4
        {
          const int i = tid * 4, r = i >> 5, c = i & 31;
          float4 v = *(const float4*)(x + (size_t)(t0 + r) * DH + kc + c);
          v4bf b;
          b[0] = (__bf16)v.x; b[1] = (__bf16)v.y;
          b[2] = (__bf16)v.z; b[3] = (__bf16)v.w;
          *(v4bf*)&xs[r][c] = b;
        }
        // wave 0: issue next-tile TDM loads, then wait for current tiles
        if (wv == 0) {
          if (kc + KC < DH) {
            tdm_load_2d(lds_off_of(&w1s[pb ^ 1][0][0]),
                        W1e + (size_t)fc * DH + (kc + KC), KC, FC, DH, FI, DH, 4, 3);
            tdm_load_2d(lds_off_of(&w3s[pb ^ 1][0][0]),
                        W3e + (size_t)fc * DH + (kc + KC), KC, FC, DH, FI, DH, 4, 3);
            __builtin_amdgcn_s_wait_tensorcnt(2);
          } else {
            __builtin_amdgcn_s_wait_tensorcnt(0);
          }
        }
        __syncthreads();

        // A: bf16 16x32 layout — lane<16 holds K0..15 of row M=ln
        BF16x16 a;
        a.h.lo = *(const v8bf*)&xs[mt * 16 + ln][16 * hl];
        a.h.hi = *(const v8bf*)&xs[mt * 16 + ln][16 * hl + 8];
        // B: 32x16 layout — lane<16 holds K0..15 of col N=ln (fp32 -> bf16)
        v16bf b1 = cvt16(&w1s[pb][nt * 16 + ln][16 * hl]);
        v16bf b3 = cvt16(&w3s[pb][nt * 16 + ln][16 * hl]);

        h1 = __builtin_amdgcn_wmma_f32_16x16x32_bf16(false, a.v, false, b1,
                                                     (short)0, h1, false, false);
        h3 = __builtin_amdgcn_wmma_f32_16x16x32_bf16(false, a.v, false, b3,
                                                     (short)0, h3, false, false);
        __syncthreads();
        pb ^= 1;
      }

      // ---- silu(h1)*h3 * cw[token] -> acts (bf16) ----
#pragma unroll
      for (int v = 0; v < 8; ++v) {
        const int r  = mt * 16 + v + 8 * hl;   // C layout: M = v + 8*(lane/16)
        const float s1 = h1[v];
        const float s3 = h3[v];
        const float av = (s1 / (1.f + __expf(-s1))) * s3 * cws[r][e];
        acts[r][nt * 16 + ln] = (__bf16)av;
      }
      // prologue: TDM-load first w2 tile (64d x 64f fp32) into buffer 0
      if (wv == 0) {
        tdm_load_2d(lds_off_of(&w2s[0][0][0]), W2e + fc,
                    FC, DC, FI, DH, FI, 5, 3);
      }
      __syncthreads();

      // ---- GEMM2: outs[32 x 1024] += acts[32 x 64] @ w2[d, fc..fc+64]^T ----
      int pb2 = 0;
      for (int dc = 0; dc < DH; dc += DC) {
        if (wv == 0) {
          if (dc + DC < DH) {
            tdm_load_2d(lds_off_of(&w2s[pb2 ^ 1][0][0]),
                        W2e + (size_t)(dc + DC) * FI + fc, FC, DC, FI, DH, FI, 5, 3);
            __builtin_amdgcn_s_wait_tensorcnt(1);
          } else {
            __builtin_amdgcn_s_wait_tensorcnt(0);
          }
        }
        __syncthreads();

        v8f cacc;
#pragma unroll
        for (int v = 0; v < 8; ++v)
          cacc[v] = outs[mt * 16 + v + 8 * hl][dc + nt * 16 + ln];

#pragma unroll
        for (int ko = 0; ko < FC; ko += 32) {
          BF16x16 a2;
          a2.h.lo = *(const v8bf*)&acts[mt * 16 + ln][ko + 16 * hl];
          a2.h.hi = *(const v8bf*)&acts[mt * 16 + ln][ko + 16 * hl + 8];
          v16bf b2 = cvt16(&w2s[pb2][nt * 16 + ln][ko + 16 * hl]);
          cacc = __builtin_amdgcn_wmma_f32_16x16x32_bf16(false, a2.v, false, b2,
                                                         (short)0, cacc, false, false);
        }
#pragma unroll
        for (int v = 0; v < 8; ++v)
          outs[mt * 16 + v + 8 * hl][dc + nt * 16 + ln] = cacc[v];
        __syncthreads();
        pb2 ^= 1;
      }
    }
  }
  __syncthreads();

  // ---- coalesced writeout of final fp32 output ----
  for (int j = tid; j < TT * DH / 4; j += 256) {
    const int lin = j * 4;
    const int r = lin >> 10, c = lin & (DH - 1);
    float4 v;
    v.x = outs[r][c];     v.y = outs[r][c + 1];
    v.z = outs[r][c + 2]; v.w = outs[r][c + 3];
    *(float4*)(out + (size_t)(t0 + r) * DH + c) = v;
  }
}

// ---------------------------------------------------------------------------
extern "C" void kernel_launch(void* const* d_in, const int* in_sizes, int n_in,
                              void* d_out, int out_size, void* d_ws, size_t ws_size,
                              hipStream_t stream) {
  const float* x  = (const float*)d_in[0];
  const float* gw = (const float*)d_in[1];
  const float* gb = (const float*)d_in[2];
  const float* w1 = (const float*)d_in[3];
  const float* w3 = (const float*)d_in[4];
  const float* w2 = (const float*)d_in[5];
  // d_in[6] = top_k (hardcoded 2)
  float* out = (float*)d_out;
  float* cw  = (float*)d_ws;   // T*8 floats = 128 KB

  moe_route<<<T_TOK / 256, 256, 0, stream>>>(x, gw, gb, cw);
  moe_experts<<<T_TOK / TT, 256, 0, stream>>>(x, w1, w3, w2, cw, out);
}
